// SoftVSAStateUpdate_19241453486177
// MI455X (gfx1250) — compile-verified, matrix operand order checked by reading, not run on hardware
//
#include <hip/hip_runtime.h>

// Problem constants (match reference setup_inputs exactly)
#define B_   8
#define L_   16384
#define D_   256
#define K_   8
#define NG_  2048          // L_/K_

typedef __attribute__((ext_vector_type(16))) __bf16 v16bf;
typedef __attribute__((ext_vector_type(8)))  float  v8f;

// ---------------------------------------------------------------------------
// Phase 1: group_vecs[b,g,:] = l2norm( mean_k( tv[b,g*8+k,:] * lp[k,:] ) )
// One block per (b,g); 256 threads = one channel each. Memory-bound: 128 MB read.
// ---------------------------------------------------------------------------
__global__ __launch_bounds__(256) void gv_kernel(const float* __restrict__ tv,
                                                 const float* __restrict__ lp,
                                                 float* __restrict__ gv) {
  const int bg = blockIdx.x;            // b*NG_ + g
  const int d  = threadIdx.x;
  const float* base = tv + (size_t)bg * (K_ * D_) + d;
  float acc = 0.0f;
#pragma unroll
  for (int k = 0; k < K_; ++k)
    acc += base[k * D_] * lp[k * D_ + d];
  const float mean = acc * (1.0f / (float)K_);

  __shared__ float red[256];
  red[d] = mean * mean;
  __syncthreads();
  for (int s = 128; s > 0; s >>= 1) {
    if (d < s) red[d] += red[d + s];
    __syncthreads();
  }
  const float inv = 1.0f / fmaxf(sqrtf(red[0]), 1e-8f);
  gv[(size_t)bg * D_ + d] = mean * inv;
}

// ---------------------------------------------------------------------------
// Phase 1b: W_q fp32 -> bf16 (stays resident in L2 for the GEMM: 128 KB)
// ---------------------------------------------------------------------------
__global__ __launch_bounds__(256) void wq_cvt_kernel(const float* __restrict__ wq,
                                                     __bf16* __restrict__ wqb) {
  const int i = blockIdx.x * 256 + threadIdx.x;   // grid = 256 blocks -> 65536
  wqb[i] = (__bf16)wq[i];
}

// ---------------------------------------------------------------------------
// Phase 2: sequential scan. One wave32 per batch; 8 channels per lane in
// registers; cross-lane reduction via shuffles (no LDS/barriers); software
// prefetch of next step's gv/mp to hide L2 latency behind the rsqrt chain.
// ---------------------------------------------------------------------------
__global__ __launch_bounds__(32) void scan_kernel(const float* __restrict__ gv,
                                                  const float* __restrict__ mp,
                                                  const float* __restrict__ dl,
                                                  float* __restrict__ gs,
                                                  float* __restrict__ final_out) {
  const int b    = blockIdx.x;
  const int lane = threadIdx.x;          // 0..31
  const int dbase = lane * 8;            // 8 contiguous channels per lane

  float alpha[8], S[8];
#pragma unroll
  for (int j = 0; j < 8; ++j) {
    const float x = dl[dbase + j];
    alpha[j] = 1.0f / (1.0f + __expf(-x));
    S[j] = 0.0f;
  }

  const float* gvb = gv + (size_t)b * NG_ * D_ + dbase;
  float*       gsb = gs + (size_t)b * NG_ * D_ + dbase;

  float g8[8], m8[8];
#pragma unroll
  for (int j = 0; j < 8; ++j) { g8[j] = gvb[j]; m8[j] = mp[dbase + j]; }

  for (int g = 0; g < NG_; ++g) {
    // prefetch next step (hides load latency behind reduction below)
    float gn[8], mn[8];
    if (g + 1 < NG_) {
      const size_t o = (size_t)(g + 1) * D_;
#pragma unroll
      for (int j = 0; j < 8; ++j) { gn[j] = gvb[o + j]; mn[j] = mp[o + dbase + j]; }
    } else {
#pragma unroll
      for (int j = 0; j < 8; ++j) { gn[j] = 0.0f; mn[j] = 0.0f; }
    }

    float v[8], ss = 0.0f;
#pragma unroll
    for (int j = 0; j < 8; ++j) {
      const float t = alpha[j] * S[j] + (1.0f - alpha[j]) * (g8[j] * m8[j]);
      v[j] = t;
      ss += t * t;
    }
    // wave32 all-reduce of sum-of-squares
    ss += __shfl_xor(ss, 16, 32);
    ss += __shfl_xor(ss,  8, 32);
    ss += __shfl_xor(ss,  4, 32);
    ss += __shfl_xor(ss,  2, 32);
    ss += __shfl_xor(ss,  1, 32);
    const float inv = 1.0f / fmaxf(sqrtf(ss), 1e-8f);

    const size_t so = (size_t)g * D_;
#pragma unroll
    for (int j = 0; j < 8; ++j) {
      S[j] = v[j] * inv;
      gsb[so + j] = S[j];
    }
#pragma unroll
    for (int j = 0; j < 8; ++j) { g8[j] = gn[j]; m8[j] = mn[j]; }
  }
  // final_state output
#pragma unroll
  for (int j = 0; j < 8; ++j) final_out[b * D_ + dbase + j] = S[j];
}

// ---------------------------------------------------------------------------
// Phase 3: queries = (state_prev ⊙ lp) @ W_q^T via bf16 WMMA, f32 accum.
// Block = 256 token rows (32 groups) x 256 cols; 8 waves, each wave owns TWO
// 16-row strips (rows wave*16.. and 128+wave*16..) so every B-tile fetched
// from LDS feeds two WMMAs. B-tiles are double-buffered in registers so the
// ds_load for tile nt+1 overlaps the WMMAs of tile nt (fewer s_wait_dscnt
// stalls on the matrix pipe).
// ---------------------------------------------------------------------------
__global__ __launch_bounds__(256) void q_kernel(const float* __restrict__ gs,
                                                const float* __restrict__ lp,
                                                const __bf16* __restrict__ wqb,
                                                float* __restrict__ out) {
  __shared__ __bf16 WqS[256][40];   // 32 used, padded: conflict-free lane stride
  __shared__ float  stS[32][257];   // prev-state per group tile, padded
  __shared__ float  lpS[8][257];    // local_positions, padded

  const int tid    = threadIdx.x;
  const int wave   = tid >> 5;
  const int lane   = tid & 31;
  const int lanelo = lane & 15;
  const int khalf  = lane >> 4;

  const int r0 = blockIdx.x * 256;     // flat (b,l) row base; blocks never straddle b
  const int b  = r0 / L_;
  const int l0 = r0 % L_;
  const int g0 = l0 >> 3;              // first of 32 groups in this block

  // Stage prev-states (zero for global group 0) and local_positions.
  for (int idx = tid; idx < 32 * D_; idx += 256) {
    const int gi = idx >> 8, d = idx & 255;
    const int gg = g0 + gi;
    stS[gi][d] = (gg == 0) ? 0.0f : gs[((size_t)b * NG_ + (gg - 1)) * D_ + d];
  }
  for (int idx = tid; idx < K_ * D_; idx += 256)
    lpS[idx >> 8][idx & 255] = lp[idx];

  const int m   = lanelo;              // row within a 16-row strip
  const int gi0 = wave * 2 + (m >> 3); // group tile for strip 0
  const int gi1 = 16 + gi0;            // group tile for strip 1 (rows +128)
  const int kk  = m & 7;               // position-in-group -> lp row

  v8f acc0[16], acc1[16];
#pragma unroll
  for (int nt = 0; nt < 16; ++nt)
#pragma unroll
    for (int j = 0; j < 8; ++j) { acc0[nt][j] = 0.0f; acc1[nt][j] = 0.0f; }

  for (int kt = 0; kt < 8; ++kt) {     // K slices of 32
    __syncthreads();
    {  // stage W_q[:, kt*32 .. kt*32+31] as bf16 (row e = tid)
      const __bf16* src = wqb + (size_t)tid * D_ + kt * 32;
#pragma unroll
      for (int dd = 0; dd < 32; ++dd) WqS[tid][dd] = src[dd];
    }
    __syncthreads();

    // Build both A fragments: X[row,d] = state_prev[row] * lp[k(row)][d] (bf16)
    v16bf a0, a1;
    {
      const float* s0 = &stS[gi0][0];
      const float* s1 = &stS[gi1][0];
      const float* lv = &lpS[kk][0];
      const int base = kt * 32 + khalf * 8;
#pragma unroll
      for (int j = 0; j < 8; ++j) {
        const float lA = lv[base + j], lB = lv[base + 16 + j];
        a0[j]     = (__bf16)(s0[base + j]      * lA);
        a0[j + 8] = (__bf16)(s0[base + 16 + j] * lB);
        a1[j]     = (__bf16)(s1[base + j]      * lA);
        a1[j + 8] = (__bf16)(s1[base + 16 + j] * lB);
      }
    }

    // Double-buffered B tiles: load nt+1 while WMMAs of nt issue.
    v16bf bt;
    {
      const __bf16* wr = &WqS[lanelo][khalf * 8];
#pragma unroll
      for (int j = 0; j < 8; ++j) { bt[j] = wr[j]; bt[j + 8] = wr[16 + j]; }
    }
#pragma unroll
    for (int nt = 0; nt < 16; ++nt) {
      v16bf btn;
      if (nt + 1 < 16) {
        const __bf16* wr = &WqS[(nt + 1) * 16 + lanelo][khalf * 8];
#pragma unroll
        for (int j = 0; j < 8; ++j) { btn[j] = wr[j]; btn[j + 8] = wr[16 + j]; }
      }
      acc0[nt] = __builtin_amdgcn_wmma_f32_16x16x32_bf16(
          false, a0, false, bt, (short)0, acc0[nt], false, false);
      acc1[nt] = __builtin_amdgcn_wmma_f32_16x16x32_bf16(
          false, a1, false, bt, (short)0, acc1[nt], false, false);
      if (nt + 1 < 16) bt = btn;
    }
  }

  // Store: C/D layout — VGPR j holds row M=j+8*khalf, col N=lanelo per N-tile.
  const size_t rb0 = (size_t)(r0 + wave * 16) * D_;
  const size_t rb1 = (size_t)(r0 + 128 + wave * 16) * D_;
#pragma unroll
  for (int nt = 0; nt < 16; ++nt)
#pragma unroll
    for (int j = 0; j < 8; ++j) {
      const int M = j + 8 * khalf;
      out[rb0 + (size_t)M * D_ + nt * 16 + lanelo] = acc0[nt][j];
      out[rb1 + (size_t)M * D_ + nt * 16 + lanelo] = acc1[nt][j];
    }
}

// ---------------------------------------------------------------------------
extern "C" void kernel_launch(void* const* d_in, const int* in_sizes, int n_in,
                              void* d_out, int out_size, void* d_ws, size_t ws_size,
                              hipStream_t stream) {
  const float* tv = (const float*)d_in[0];   // (B,L,d)
  const float* lp = (const float*)d_in[1];   // (K,d)
  const float* mp = (const float*)d_in[2];   // (NG,d)
  const float* dl = (const float*)d_in[3];   // (d,)
  const float* wq = (const float*)d_in[4];   // (d,d)
  float* out = (float*)d_out;                // queries (B,L,d) ++ final_state (B,d)

  // workspace layout: gv (16MB) | gs (16MB) | wq_bf16 (128KB)
  float*  gv  = (float*)d_ws;
  float*  gs  = gv + (size_t)B_ * NG_ * D_;
  __bf16* wqb = (__bf16*)((char*)d_ws + 2 * (size_t)B_ * NG_ * D_ * sizeof(float));

  gv_kernel    <<<B_ * NG_,        256, 0, stream>>>(tv, lp, gv);
  wq_cvt_kernel<<<D_,              256, 0, stream>>>(wq, wqb);
  scan_kernel  <<<B_,               32, 0, stream>>>(gv, mp, dl, gs,
                                                     out + (size_t)B_ * L_ * D_);
  q_kernel     <<<(B_ * L_) / 256, 256, 0, stream>>>(gs, lp, wqb, out);
}
